// TopKGating_40235253629367
// MI455X (gfx1250) — compile-verified, matrix operand order checked by reading, not run on hardware
//
#include <hip/hip_runtime.h>
#include <hip/hip_bf16.h>

#define TOKENS 32768   // 4 * 8192
#define HDIM   2048
#define NEXP   64
#define KC     256                  // K-chunk staged in LDS
#define BROW   (KC + 8)             // padded LDS row stride (bf16 elems) -> conflict-free
#define LROW   68                   // padded logit row stride (floats)

typedef __attribute__((ext_vector_type(16))) __bf16 v16bf;
typedef __attribute__((ext_vector_type(8)))  float  v8f;
typedef __attribute__((ext_vector_type(4)))  float  f32x4;
typedef __attribute__((ext_vector_type(4)))  unsigned int u32x4;

union BFrag {
    v16bf v;
    unsigned short s[16];
    u32x4 q[2];
};

__device__ __forceinline__ unsigned short f2bf_trunc(float f) {
    union { float f; unsigned u; } c; c.f = f;
    return (unsigned short)(c.u >> 16);
}
__device__ __forceinline__ float bf2f(unsigned short h) {
    union { unsigned u; float f; } c; c.u = ((unsigned)h) << 16;
    return c.f;
}

// ---------------------------------------------------------------------------
// Prep: split W (fp32) into bf16 hi/lo planes; zero the global accumulators.
// ---------------------------------------------------------------------------
__global__ void moe_prep_kernel(const float* __restrict__ W,
                                unsigned short* __restrict__ Whi,
                                unsigned short* __restrict__ Wlo,
                                float* __restrict__ gAcc) {
    int i = blockIdx.x * 256 + threadIdx.x;
    if (i < NEXP * HDIM) {
        float w = W[i];
        unsigned short hi = f2bf_trunc(w);
        Whi[i] = hi;
        Wlo[i] = f2bf_trunc(w - bf2f(hi));
    }
    if (i < 2 * NEXP) gAcc[i] = 0.0f;   // [0..63]=sum(probs), [64..127]=count(probs>0)
}

// ---------------------------------------------------------------------------
// Main: per wave, 16 tokens x 64 experts via split-bf16 WMMA, then top-2 +
// softmax + aux-loss accumulation.
// ---------------------------------------------------------------------------
__global__ __launch_bounds__(256) void moe_gate_kernel(
    const float* __restrict__ x,
    const unsigned short* __restrict__ Whi,
    const unsigned short* __restrict__ Wlo,
    float* __restrict__ gatesOut,   // [TOKENS][2]
    float* __restrict__ idxOut,     // [TOKENS][2] (indices as float)
    float* __restrict__ gAcc)       // [128] global accumulators
{
    __shared__ unsigned short sHi[NEXP * BROW];
    __shared__ unsigned short sLo[NEXP * BROW];
    __shared__ float sLog[8][16][LROW];
    __shared__ float eSum[NEXP];
    __shared__ float eCnt[NEXP];

    const int tid    = threadIdx.x;
    const int wv     = tid >> 5;
    const int lane   = tid & 31;
    const int lhalf  = lane >> 4;      // 0: K+0 half, 1: K+8 half
    const int l16    = lane & 15;
    const int rowBase = blockIdx.x * 128 + wv * 16;

    if (tid < NEXP) { eSum[tid] = 0.0f; eCnt[tid] = 0.0f; }

    v8f acc[4];
    #pragma unroll
    for (int t = 0; t < 4; ++t) acc[t] = (v8f){0.f,0.f,0.f,0.f,0.f,0.f,0.f,0.f};

    const float* xrow = x + (size_t)(rowBase + l16) * HDIM;

    for (int kc = 0; kc < HDIM; kc += KC) {
        __syncthreads();
        // Cooperative stage of W_hi/W_lo K-chunk into padded LDS (16B moves).
        for (int idx = tid; idx < NEXP * (KC / 8); idx += 256) {
            int e   = idx >> 5;          // KC/8 = 32 segments per expert row
            int seg = idx & 31;
            *(u32x4*)(sHi + e * BROW + seg * 8) =
                *(const u32x4*)(Whi + e * HDIM + kc + seg * 8);
            *(u32x4*)(sLo + e * BROW + seg * 8) =
                *(const u32x4*)(Wlo + e * HDIM + kc + seg * 8);
        }
        __syncthreads();

        for (int ks = 0; ks < KC; ks += 32) {
            const int k0 = kc + ks + lhalf * 8;
            // A fragment: rows on lanes, K runs [k0..k0+7] and [k0+16..k0+23]
            f32x4 f0 = *(const f32x4*)(xrow + k0);
            f32x4 f1 = *(const f32x4*)(xrow + k0 + 4);
            f32x4 f2 = *(const f32x4*)(xrow + k0 + 16);
            f32x4 f3 = *(const f32x4*)(xrow + k0 + 20);
            float fv[16] = { f0.x, f0.y, f0.z, f0.w,  f1.x, f1.y, f1.z, f1.w,
                             f2.x, f2.y, f2.z, f2.w,  f3.x, f3.y, f3.z, f3.w };
            BFrag ah, al;
            #pragma unroll
            for (int i = 0; i < 16; ++i) {
                unsigned short h = f2bf_trunc(fv[i]);
                ah.s[i] = h;
                al.s[i] = f2bf_trunc(fv[i] - bf2f(h));
            }

            const int boff = ks + lhalf * 8;
            #pragma unroll
            for (int t = 0; t < 4; ++t) {
                const int e = t * 16 + l16;       // expert column on lanes
                const unsigned short* hp = sHi + e * BROW + boff;
                const unsigned short* lp = sLo + e * BROW + boff;
                BFrag bh, bl;
                bh.q[0] = *(const u32x4*)(hp);
                bh.q[1] = *(const u32x4*)(hp + 16);
                bl.q[0] = *(const u32x4*)(lp);
                bl.q[1] = *(const u32x4*)(lp + 16);
                // split-bf16: x_hi*W_hi + x_lo*W_hi + x_hi*W_lo (fp32 accum)
                acc[t] = __builtin_amdgcn_wmma_f32_16x16x32_bf16(
                             false, ah.v, false, bh.v, (short)0, acc[t], false, false);
                acc[t] = __builtin_amdgcn_wmma_f32_16x16x32_bf16(
                             false, al.v, false, bh.v, (short)0, acc[t], false, false);
                acc[t] = __builtin_amdgcn_wmma_f32_16x16x32_bf16(
                             false, ah.v, false, bl.v, (short)0, acc[t], false, false);
            }
        }
    }

    // Spill logits to LDS: C/D layout -> lane holds col (l16+16t), rows r(+8*lhalf).
    #pragma unroll
    for (int t = 0; t < 4; ++t)
        #pragma unroll
        for (int r = 0; r < 8; ++r)
            sLog[wv][r + lhalf * 8][t * 16 + l16] = acc[t][r];

    __syncthreads();

    if (lane < 16) {
        const int token = rowBase + l16;
        const float* lg = &sLog[wv][l16][0];

        float v1 = -__builtin_inff(), v2 = -__builtin_inff();
        int   i1 = 0, i2 = 0;
        for (int c = 0; c < NEXP; ++c) {          // strict > == first-index tiebreak
            float v = lg[c];
            if (v > v1)      { v2 = v1; i2 = i1; v1 = v; i1 = c; }
            else if (v > v2) { v2 = v;  i2 = c; }
        }
        // softmax over the top-2 (v1 >= v2)
        float e2 = __expf(v2 - v1);
        float g1 = 1.0f / (1.0f + e2);
        gatesOut[token * 2 + 0] = g1;
        gatesOut[token * 2 + 1] = e2 * g1;
        idxOut[token * 2 + 0] = (float)i1;
        idxOut[token * 2 + 1] = (float)i2;

        // full softmax for load-balance loss
        float Z = 0.0f;
        for (int c = 0; c < NEXP; ++c) Z += __expf(lg[c] - v1);
        float invZ = 1.0f / Z;
        for (int c = 0; c < NEXP; ++c) {
            float p = __expf(lg[c] - v1) * invZ;
            atomicAdd(&eSum[c], p);
            atomicAdd(&eCnt[c], (p > 0.0f) ? 1.0f : 0.0f);
        }
    }

    __syncthreads();
    if (tid < NEXP) {
        atomicAdd(&gAcc[tid], eSum[tid]);
        atomicAdd(&gAcc[NEXP + tid], eCnt[tid]);
    }
}

// ---------------------------------------------------------------------------
// Finalize: loss = E * sum_e (sum_e/T) * (cnt_e/T)
// ---------------------------------------------------------------------------
__global__ void moe_loss_kernel(const float* __restrict__ gAcc,
                                float* __restrict__ loss) {
    __shared__ float red[NEXP];
    const int e = threadIdx.x;
    const float invT = 1.0f / (float)TOKENS;
    red[e] = (gAcc[e] * invT) * (gAcc[NEXP + e] * invT);
    __syncthreads();
    for (int s = 32; s > 0; s >>= 1) {
        if (e < s) red[e] += red[e + s];
        __syncthreads();
    }
    if (e == 0) loss[0] = (float)NEXP * red[0];
}

extern "C" void kernel_launch(void* const* d_in, const int* in_sizes, int n_in,
                              void* d_out, int out_size, void* d_ws, size_t ws_size,
                              hipStream_t stream) {
    const float* x = (const float*)d_in[0];      // (4,8192,2048) fp32
    const float* W = (const float*)d_in[1];      // (64,2048) fp32
    float* out = (float*)d_out;                  // gates(65536) | idx(65536) | loss(1)

    unsigned short* Whi  = (unsigned short*)d_ws;
    unsigned short* Wlo  = Whi + NEXP * HDIM;
    float*          gAcc = (float*)(Wlo + NEXP * HDIM);

    moe_prep_kernel<<<(NEXP * HDIM + 255) / 256, 256, 0, stream>>>(W, Whi, Wlo, gAcc);
    moe_gate_kernel<<<TOKENS / 128, 256, 0, stream>>>(
        x, Whi, Wlo, out, out + 2 * TOKENS, gAcc);
    moe_loss_kernel<<<1, NEXP, 0, stream>>>(gAcc, out + 4 * TOKENS);
}
